// ConvNMF_5609227288997
// MI455X (gfx1250) — matrix operand order repaired and checked by reading, not compiled, not run on hardware
//
#include <hip/hip_runtime.h>
#include <hip/hip_fp16.h>

typedef __attribute__((ext_vector_type(16))) _Float16 v16h;
typedef __attribute__((ext_vector_type(8)))  float    v8f;

#define NF   256            // n_features (M)
#define NC   64             // n_components
#define NL   32             // n_lags
#define NT   20000          // n_timebins (N)
#define KTOT 2048           // NC*NL (K)
#define NKB  64             // KTOT / 32 (k-blocks per WMMA chain)
#define HROW 20031          // NT + NL - 1
#define HSTR 20032          // padded H row stride (even -> aligned row starts)

// ---------------------------------------------------------------------------
// softplus, numerically stable: max(x,0) + log1p(exp(-|x|))
__device__ __forceinline__ float softplusf(float x) {
    return fmaxf(x, 0.0f) + log1pf(expf(-fabsf(x)));
}

// ---------------------------------------------------------------------------
// Kernel 1: softplus(W_pre) -> f16, pre-swizzled into WMMA A-fragment layout.
// A is 256x2048 row-major (flat W_pre). Fragment (tileM, kb): contiguous-k
// 16-bit A 16x32 layout: lane 0-15 = row M, element j of v16h maps to
// K = j + 16*(lane>=16).
// Aswz (as v16h array) index = (tileM*64 + kb)*32 + lane  -> aligned 32B loads.
__global__ __launch_bounds__(256) void stage_A(const float* __restrict__ W_pre,
                                               _Float16* __restrict__ Aswz) {
    int idx  = blockIdx.x * 256 + threadIdx.x;      // 16*64*32*16 = 524288
    int j    = idx & 15;
    int lane = (idx >> 4) & 31;
    int kb   = (idx >> 9) & 63;
    int tm   = idx >> 15;
    int f    = tm * 16 + (lane & 15);
    int K    = kb * 32 + j + ((lane >> 4) << 4);
    Aswz[idx] = (_Float16)softplusf(W_pre[f * KTOT + K]);
}

// ---------------------------------------------------------------------------
// Kernel 2: softplus(H_pre) -> f16, row-padded to stride HSTR.
__global__ __launch_bounds__(256) void stage_H(const float* __restrict__ H_pre,
                                               _Float16* __restrict__ Hf16) {
    int idx = blockIdx.x * 256 + threadIdx.x;       // 64*20032 = 1282048
    int k   = idx / HSTR;
    int i   = idx - k * HSTR;
    float sp = 0.0f;
    if (i < HROW) sp = softplusf(H_pre[k * HROW + i]);
    Hf16[idx] = (_Float16)sp;
}

// ---------------------------------------------------------------------------
// Kernel 3: fused GEMM (WMMA f16 -> f32) + squared-residual reduction.
// One block per 16-wide t-strip (1250 blocks). 8 waves; wave w computes row
// tiles {w, w+8}; B fragments shared between the two WMMAs.
// Hwin[k,l,t] = H[k, 19999 + l - t]; for this strip the needed window per k
// is 47 contiguous values -> staged in LDS.
__global__ __launch_bounds__(256) void convnmf_gemm(
        const _Float16* __restrict__ Aswz,
        const _Float16* __restrict__ Hf16,
        const float*    __restrict__ data,
        double*         __restrict__ acc) {
    __shared__ _Float16 hh[NC * 48];   // 47 used per k, padded to 48
    __shared__ float    wsum[8];

    const int tileN = blockIdx.x;
    const int t0    = tileN * 16;
    const int ibase = (NT - 16) - t0;          // lowest H index needed
    const int tid   = threadIdx.x;

    // cooperative window stage: 64 rows x 47 halves
    for (int e = tid; e < NC * 47; e += 256) {
        int k   = e / 47;
        int off = e - k * 47;
        hh[k * 48 + off] = Hf16[k * HSTR + ibase + off];
    }
    __syncthreads();

    const int wave    = tid >> 5;
    const int lane    = tid & 31;
    const int n       = lane & 15;             // N within tile
    const int laneoff = (lane >> 4) << 4;      // 0 or 16 (K offset for B)
    const int tm0     = wave;
    const int tm1     = wave + 8;

    const v16h* __restrict__ Abase = (const v16h*)Aswz;
    v8f c0 = {};
    v8f c1 = {};

    // B fragment element j (f16): l = laneoff + j, value at hh offset
    // (19999 + l - (t0+n)) - ibase = 15 + laneoff + j - n  -> 16 contiguous.
    const _Float16* __restrict__ hrow0 = &hh[15 + laneoff - n];

    for (int kb = 0; kb < NKB; ++kb) {
        const _Float16* hp = hrow0 + kb * 48;
        v16h b;
#pragma unroll
        for (int j = 0; j < 16; ++j) b[j] = hp[j];

        v16h a0 = Abase[(tm0 * NKB + kb) * 32 + lane];
        v16h a1 = Abase[(tm1 * NKB + kb) * 32 + lane];

        c0 = __builtin_amdgcn_wmma_f32_16x16x32_f16(
                 false, a0, false, b, (short)0, c0, false, false);
        c1 = __builtin_amdgcn_wmma_f32_16x16x32_f16(
                 false, a1, false, b, (short)0, c1, false, false);
    }

    // Fused residual: C/D layout -> VGPR r holds M = r + 8*(lane>=16), N = lane&15
    const int trow = t0 + n;
    const int mofs = (lane >> 4) << 3;
    float s = 0.0f;
#pragma unroll
    for (int r = 0; r < 8; ++r) {
        int   f0 = tm0 * 16 + r + mofs;
        float d0 = c0[r] - data[f0 * NT + trow];
        s += d0 * d0;
        int   f1 = tm1 * 16 + r + mofs;
        float d1 = c1[r] - data[f1 * NT + trow];
        s += d1 * d1;
    }
    // wave32 reduce
#pragma unroll
    for (int off = 16; off > 0; off >>= 1) s += __shfl_down(s, off, 32);
    if (lane == 0) wsum[wave] = s;
    __syncthreads();
    if (tid == 0) {
        float tot = 0.0f;
#pragma unroll
        for (int w = 0; w < 8; ++w) tot += wsum[w];
        atomicAdd(acc, (double)tot);   // global_atomic_add_f64
    }
}

// ---------------------------------------------------------------------------
__global__ void finalize(const double* __restrict__ acc, float* __restrict__ out) {
    out[0] = (float)(acc[0] * (1.0 / ((double)NF * (double)NT)));
}

// ---------------------------------------------------------------------------
extern "C" void kernel_launch(void* const* d_in, const int* in_sizes, int n_in,
                              void* d_out, int out_size, void* d_ws, size_t ws_size,
                              hipStream_t stream) {
    const float* data  = (const float*)d_in[0];   // (256, 20000)
    const float* W_pre = (const float*)d_in[1];   // (256, 64, 32)
    const float* H_pre = (const float*)d_in[2];   // (64, 20031)
    float*       out   = (float*)d_out;

    char*     ws   = (char*)d_ws;
    double*   acc  = (double*)ws;                         // 8 B (zeroed below)
    _Float16* Aswz = (_Float16*)(ws + 128);               // 524288 halves (1 MB)
    _Float16* Hf16 = (_Float16*)(ws + 128 + 524288 * 2);  // 1282048 halves (2.56 MB)

    hipMemsetAsync(acc, 0, sizeof(double), stream);

    stage_A<<<524288 / 256, 256, 0, stream>>>(W_pre, Aswz);
    stage_H<<<(NC * HSTR) / 256, 256, 0, stream>>>(H_pre, Hf16);
    convnmf_gemm<<<NT / 16, 256, 0, stream>>>(Aswz, Hf16, data, acc);
    finalize<<<1, 1, 0, stream>>>(acc, out);
}
